// BlockResMLP_MixerBlock_27504970564331
// MI455X (gfx1250) — matrix operand order repaired and checked by reading, compiled, not run on hardware
//
#include <hip/hip_runtime.h>

// ---------------------------------------------------------------------------
// BlockResMLP mixer for MI455X (gfx1250), bf16 WMMA, fp32 accumulate/residual.
// Async global<->LDS staging; lane-major pre-swizzled weights; inline-0 C
// operands with bias folded into ELU/residual adds.
// ---------------------------------------------------------------------------

typedef __attribute__((ext_vector_type(16))) __bf16 v16bf;
typedef __attribute__((ext_vector_type(8)))  float  v8f;

#define N_BLOCKS  32
#define BLOCK_DIM 32
#define HIDDEN    64
#define INPUT_DIM 1024
#define BT        16      // batch rows per workgroup (WMMA M)
#define S_ROW     1028    // padded LDS row stride in floats (bank-conflict-free)
#define H_ROW     72      // H scratch row stride in ushorts (16B align, no conflicts)
#define W1_FRAGS  32768   // u32 fragments per layer for W1
#define WS_LAYER  65536   // u32 per layer (W1 + W2 fragments)

#if __has_builtin(__builtin_amdgcn_global_load_async_to_lds_b128) && \
    __has_builtin(__builtin_amdgcn_global_store_async_from_lds_b128) && \
    __has_builtin(__builtin_amdgcn_s_wait_asynccnt)
#define USE_ASYNC_LDS 1
typedef __attribute__((ext_vector_type(4))) int v4i;
typedef __attribute__((address_space(1))) v4i GV4;   // global int4*
typedef __attribute__((address_space(3))) v4i LV4;   // LDS int4*
#else
#define USE_ASYNC_LDS 0
#endif

struct U8 { unsigned u[8]; };

static __device__ __forceinline__ unsigned pk_bf16(float lo, float hi) {
#if __has_builtin(__builtin_amdgcn_cvt_pk_bf16_f32)
  typedef __attribute__((ext_vector_type(2))) __bf16 v2bf;
  v2bf r = __builtin_amdgcn_cvt_pk_bf16_f32(lo, hi);
  return __builtin_bit_cast(unsigned, r);
#else
  unsigned a = __float_as_uint(lo), b = __float_as_uint(hi);
  a = (a + 0x7FFFu + ((a >> 16) & 1u)) >> 16;           // RNE to bf16
  b = (b + 0x7FFFu + ((b >> 16) & 1u)) >> 16;
  return (a & 0xFFFFu) | ((b & 0xFFFFu) << 16);
#endif
}

static __device__ __forceinline__ float elu(float h) {
  return h > 0.0f ? h : (__expf(h) - 1.0f);
}

// Pre-swizzle fp32 weights into bf16 WMMA B-operand fragments, lane-major:
// frag layout [..fragment..][lane][j] so each lane loads 2x b128.
// B (32x16): lane l -> column N = l%16; dword j packs
// (K0 = 16*(j>=4) + 8*(l>=16) + 2*(j&3)) [lo], K0+1 [hi].
__global__ __launch_bounds__(256) void prep_weights(
    const float* __restrict__ W1a, const float* __restrict__ W2a,
    const float* __restrict__ W1b, const float* __restrict__ W2b,
    unsigned* __restrict__ ws)
{
  int g = blockIdx.x * 256 + threadIdx.x;
  if (g >= 2 * WS_LAYER) return;
  int layer = g / WS_LAYER;
  int r = g - layer * WS_LAYER;
  const float* W1 = layer ? W1b : W1a;
  const float* W2 = layer ? W2b : W2a;
  unsigned v;
  if (r < W1_FRAGS) {
    // idx = (((n*4 + t)*32) + lane)*8 + j   (t: N-tile of HIDDEN)
    int j = r & 7, lane = (r >> 3) & 31, t = (r >> 8) & 3, n = r >> 10;
    int k0  = ((j >> 2) << 4) + ((lane >> 4) << 3) + ((j & 3) << 1);
    int col = t * 16 + (lane & 15);
    v = pk_bf16(W1[(n * 32 + k0) * HIDDEN + col],
                W1[(n * 32 + k0 + 1) * HIDDEN + col]);
  } else {
    // idx = ((((n*2 + t2)*2 + c)*32) + lane)*8 + j  (t2: N-tile, c: K-chunk)
    int i = r - W1_FRAGS;
    int j = i & 7, lane = (i >> 3) & 31, c = (i >> 8) & 1, t2 = (i >> 9) & 1, n = i >> 10;
    int k0  = c * 32 + ((j >> 2) << 4) + ((lane >> 4) << 3) + ((j & 3) << 1);
    int col = t2 * 16 + (lane & 15);
    v = pk_bf16(W2[(n * HIDDEN + k0) * BLOCK_DIM + col],
                W2[(n * HIDDEN + k0 + 1) * BLOCK_DIM + col]);
  }
  ws[g] = v;
}

__global__ __launch_bounds__(256) void mixer_kernel(
    const float* __restrict__ x,
    const float* __restrict__ b1a, const float* __restrict__ b2a,
    const float* __restrict__ b1b, const float* __restrict__ b2b,
    const unsigned* __restrict__ wfrag,
    float* __restrict__ out)
{
  __shared__ __align__(16) float          tile[BT * S_ROW];          // 65.8 KB
  __shared__ __align__(16) unsigned short hbuf[8 * BT * H_ROW];      // 18.4 KB

  const int tid  = threadIdx.x;
  const int lane = tid & 31;
  const int wave = tid >> 5;
  const int row0 = blockIdx.x * BT;
  const int m    = lane & 15;       // WMMA row for A / column for B-C-D
  const int hi   = lane >> 4;       // lane half
  const int s8   = hi * 8;          // A-layout K offset per lane half

  const v8f czero = {};             // inline-0 C operand (SRC2 = 0)

  // ---- stage 16x1024 fp32 tile into LDS ----
  // Each thread owns one float4 column slot per row: 256 thr * 16B = 1 row.
  const float* xrow = x + (size_t)row0 * INPUT_DIM;
#if USE_ASYNC_LDS
  #pragma unroll
  for (int it = 0; it < 16; ++it) {
    __builtin_amdgcn_global_load_async_to_lds_b128(
        (GV4*)(xrow + (size_t)it * INPUT_DIM + tid * 4),
        (LV4*)(tile + it * S_ROW + tid * 4),
        0, 0);
  }
  __builtin_amdgcn_s_wait_asynccnt(0);
#else
  #pragma unroll
  for (int it = 0; it < 16; ++it) {
    *(float4*)(tile + it * S_ROW + tid * 4) =
        *(const float4*)(xrow + (size_t)it * INPUT_DIM + tid * 4);
  }
#endif
  __syncthreads();

  unsigned short* hb = hbuf + wave * (BT * H_ROW);

  #pragma unroll
  for (int layer = 0; layer < 2; ++layer) {
    const unsigned* wf1 = wfrag + layer * WS_LAYER;
    const unsigned* wf2 = wf1 + W1_FRAGS;
    const float* b1 = layer ? b1b : b1a;
    const float* b2 = layer ? b2b : b2a;

    #pragma unroll
    for (int q = 0; q < 4; ++q) {
      const int n = wave * 4 + q;             // block owned by this wave

      // ---- A fragment: 16x32 bf16. Lane row m; dword g holds K pair
      //      k0 = (g>=4)*16 + 8*hi + 2*(g&3).
      U8 au;
      if (layer == 0) {
        // contiguous block: columns [32n, 32n+32)
        const float* rp = tile + m * S_ROW + n * BLOCK_DIM;
        float4 f0 = *(const float4*)(rp + s8);
        float4 f1 = *(const float4*)(rp + s8 + 4);
        float4 f2 = *(const float4*)(rp + 16 + s8);
        float4 f3 = *(const float4*)(rp + 16 + s8 + 4);
        au.u[0] = pk_bf16(f0.x, f0.y); au.u[1] = pk_bf16(f0.z, f0.w);
        au.u[2] = pk_bf16(f1.x, f1.y); au.u[3] = pk_bf16(f1.z, f1.w);
        au.u[4] = pk_bf16(f2.x, f2.y); au.u[5] = pk_bf16(f2.z, f2.w);
        au.u[6] = pk_bf16(f3.x, f3.y); au.u[7] = pk_bf16(f3.z, f3.w);
      } else {
        // strided block: element k lives at column k*32 + n
        const float* rp = tile + m * S_ROW + n;
        #pragma unroll
        for (int g = 0; g < 8; ++g) {
          int k0 = ((g >> 2) << 4) + s8 + ((g & 3) << 1);
          au.u[g] = pk_bf16(rp[k0 * 32], rp[(k0 + 1) * 32]);
        }
      }
      v16bf A = __builtin_bit_cast(v16bf, au);

      // ---- GEMM1: H(16x64) = A @ W1 (C = inline 0; bias folded below) ----
      v8f acc1[4];
      #pragma unroll
      for (int t = 0; t < 4; ++t) {
        const uint4* wp = (const uint4*)(wf1 + (((n * 4 + t) * 32 + lane) << 3));
        uint4 qa = wp[0], qb = wp[1];
        U8 bu;
        bu.u[0] = qa.x; bu.u[1] = qa.y; bu.u[2] = qa.z; bu.u[3] = qa.w;
        bu.u[4] = qb.x; bu.u[5] = qb.y; bu.u[6] = qb.z; bu.u[7] = qb.w;
        v16bf B = __builtin_bit_cast(v16bf, bu);
        acc1[t] = __builtin_amdgcn_wmma_f32_16x16x32_bf16(
            false, A, false, B, (short)0, czero, false, false);
      }

      // ---- bias + ELU + bf16, row-pair packed, C/D-layout -> LDS scratch ----
      #pragma unroll
      for (int t = 0; t < 4; ++t) {
        const float bvt = b1[n * HIDDEN + t * 16 + m];
        const int   col = t * 16 + m;
        #pragma unroll
        for (int i = 0; i < 4; ++i) {
          float h0 = elu(acc1[t][2 * i]     + bvt);
          float h1 = elu(acc1[t][2 * i + 1] + bvt);
          unsigned p = pk_bf16(h0, h1);
          hb[(2 * i +     8 * hi) * H_ROW + col] = (unsigned short)p;
          hb[(2 * i + 1 + 8 * hi) * H_ROW + col] = (unsigned short)(p >> 16);
        }
      }
      // Same-wave LDS ops are in-order (DScnt); stop compiler reordering.
      asm volatile("" ::: "memory");

      // ---- re-gather H as two 16x32 A fragments (K-chunks of 64) ----
      v16bf A2[2];
      #pragma unroll
      for (int c2 = 0; c2 < 2; ++c2) {
        const unsigned short* hp = hb + m * H_ROW + c2 * 32;
        uint4 q0 = *(const uint4*)(hp + s8);        // K pairs for g=0..3
        uint4 q1 = *(const uint4*)(hp + 16 + s8);   // K pairs for g=4..7
        U8 a2u;
        a2u.u[0] = q0.x; a2u.u[1] = q0.y; a2u.u[2] = q0.z; a2u.u[3] = q0.w;
        a2u.u[4] = q1.x; a2u.u[5] = q1.y; a2u.u[6] = q1.z; a2u.u[7] = q1.w;
        A2[c2] = __builtin_bit_cast(v16bf, a2u);
      }

      // ---- GEMM2: Y(16x32) = H @ W2 (C=0 then chained; bias in residual) ----
      v8f acc2[2];
      #pragma unroll
      for (int t2 = 0; t2 < 2; ++t2) {
        v8f cc = czero;
        #pragma unroll
        for (int c2 = 0; c2 < 2; ++c2) {
          const uint4* wp =
              (const uint4*)(wf2 + ((((n * 2 + t2) * 2 + c2) * 32 + lane) << 3));
          uint4 qa = wp[0], qb = wp[1];
          U8 bu;
          bu.u[0] = qa.x; bu.u[1] = qa.y; bu.u[2] = qa.z; bu.u[3] = qa.w;
          bu.u[4] = qb.x; bu.u[5] = qb.y; bu.u[6] = qb.z; bu.u[7] = qb.w;
          v16bf B = __builtin_bit_cast(v16bf, bu);
          cc = __builtin_amdgcn_wmma_f32_16x16x32_bf16(
              false, A2[c2], false, B, (short)0, cc, false, false);
        }
        acc2[t2] = cc;
      }

      // ---- bias + fp32 residual + writeback (layer 1 transposed back) ----
      #pragma unroll
      for (int t2 = 0; t2 < 2; ++t2) {
        const float bv2 = b2[n * BLOCK_DIM + t2 * 16 + m];
        #pragma unroll
        for (int r = 0; r < 8; ++r) {
          int Mr   = r + 8 * hi;
          int jcol = t2 * 16 + m;               // index within block
          int addr = (layer == 0)
                       ? (Mr * S_ROW + n * BLOCK_DIM + jcol)
                       : (Mr * S_ROW + jcol * BLOCK_DIM + n);
          tile[addr] = acc2[t2][r] + (bv2 + tile[addr]);
        }
      }
    }
    __syncthreads();   // cross-block mixing boundary between layers
  }

  // ---- writeout of the final tile ----
  float* orow = out + (size_t)row0 * INPUT_DIM;
#if USE_ASYNC_LDS
  #pragma unroll
  for (int it = 0; it < 16; ++it) {
    __builtin_amdgcn_global_store_async_from_lds_b128(
        (GV4*)(orow + (size_t)it * INPUT_DIM + tid * 4),
        (LV4*)(tile + it * S_ROW + tid * 4),
        0, 0);
  }
  // S_ENDPGM performs an implicit wait-idle; no explicit ASYNCcnt wait needed.
#else
  #pragma unroll
  for (int it = 0; it < 16; ++it) {
    *(float4*)(orow + (size_t)it * INPUT_DIM + tid * 4) =
        *(const float4*)(tile + it * S_ROW + tid * 4);
  }
#endif
}

extern "C" void kernel_launch(void* const* d_in, const int* in_sizes, int n_in,
                              void* d_out, int out_size, void* d_ws, size_t ws_size,
                              hipStream_t stream) {
  const float* x   = (const float*)d_in[0];
  const float* W1a = (const float*)d_in[1];
  const float* b1a = (const float*)d_in[2];
  const float* W2a = (const float*)d_in[3];
  const float* b2a = (const float*)d_in[4];
  const float* W1b = (const float*)d_in[5];
  const float* b1b = (const float*)d_in[6];
  const float* W2b = (const float*)d_in[7];
  const float* b2b = (const float*)d_in[8];
  float*    out = (float*)d_out;
  unsigned* ws  = (unsigned*)d_ws;     // needs 2*WS_LAYER*4 = 512 KB

  const int bs = in_sizes[0] / INPUT_DIM;   // 16384

  prep_weights<<<(2 * WS_LAYER + 255) / 256, 256, 0, stream>>>(W1a, W2a, W1b, W2b, ws);
  mixer_kernel<<<bs / BT, 256, 0, stream>>>(x, b1a, b2a, b1b, b2b, ws, out);
}
